// PointPairwiseRelation3_43078521979326
// MI455X (gfx1250) — compile-verified, matrix operand order checked by reading, not run on hardware
//
#include <hip/hip_runtime.h>
#include <hip/hip_bf16.h>

// PointPairwiseRelation3 on MI455X (gfx1250, wave32).
//
// Compute-bound: ~4.3 GFLOP of per-pair 16x16 matmuls over 8.4M (j,k)
// pairs, <64 KB global traffic. Per (b,n) block: fold a, b1 and the x-terms
// into u[j][16] (waves 0-3) and v[k][16] (waves 4-7), stored in LDS as f16.
// Main loop: A = 16 pairs x 16 ch of relu(u_j+v_k) built with packed f16
// ALU, one V_WMMA_F32_16X16X32_F16 per 16 pairs (K zero-padded 16->32),
// B = W2 (f16), C = b2 broadcast. relu#2 folds into dual max/sum
// accumulators reduced with log-depth trees (no serial cndmask chain).

typedef __attribute__((ext_vector_type(16))) _Float16 v16h;
typedef __attribute__((ext_vector_type(8)))  _Float16 v8h;
typedef __attribute__((ext_vector_type(8)))  float    v8f;

#define NC   16   // C == Co
#define NPTS 256  // N
#define NJ   128  // N1
#define NK   128  // N2

__global__ __launch_bounds__(256)
void pppr_kernel(const float* __restrict__ x,    // [B,N,16]
                 const float* __restrict__ x1,   // [B,128,16]
                 const float* __restrict__ x2,   // [B,128,16]
                 const float* __restrict__ W1,   // [48,16]
                 const float* __restrict__ b1,   // [16]
                 const float* __restrict__ W2,   // [16,16]
                 const float* __restrict__ b2,   // [16]
                 float* __restrict__ out)        // [B,N,16]
{
    __shared__ __align__(16) _Float16 uS[NJ * NC];   // 4 KB
    __shared__ __align__(16) _Float16 vS[NK * NC];   // 4 KB
    __shared__ __align__(16) float    w1S[48 * NC];  // 3 KB
    __shared__ float redS[8 * NC];

    const int t   = threadIdx.x;     // 0..255
    const int blk = blockIdx.x;      // b*N + n
    const int b   = blk >> 8;        // N == 256

    for (int i = t; i < 48 * NC; i += 256) w1S[i] = W1[i];
    __syncthreads();

    // ---------------- phase 1: build u rows (waves 0-3) / v rows (waves 4-7)
    {
        const int tr = t & 127;
        float xl[NC];
        const float* xp = x + blk * NC;
        #pragma unroll
        for (int c = 0; c < NC; ++c) xl[c] = xp[c];

        const float* sp = (t < 128) ? (x1 + (b * NJ + tr) * NC)
                                    : (x2 + (b * NK + tr) * NC);
        float dd[NC];
        #pragma unroll
        for (int c = 0; c < NC; ++c) dd[c] = sp[c] - xl[c];

        float av[NC];
        if (t < 128) {
            // u[j][c] = b1[c] + (x @ Wa)[c] + ((x1[j]-x) @ Wb)[c]
            #pragma unroll
            for (int c = 0; c < NC; ++c) av[c] = b1[c];
            for (int cc = 0; cc < NC; ++cc) {
                #pragma unroll
                for (int c = 0; c < NC; ++c) {
                    av[c] = fmaf(xl[cc], w1S[cc * NC + c],        av[c]);
                    av[c] = fmaf(dd[cc], w1S[(NC + cc) * NC + c], av[c]);
                }
            }
            #pragma unroll
            for (int c = 0; c < NC; ++c) uS[tr * NC + c] = (_Float16)av[c];
        } else {
            // v[k][c] = ((x2[k]-x) @ Wc)[c]
            #pragma unroll
            for (int c = 0; c < NC; ++c) av[c] = 0.0f;
            for (int cc = 0; cc < NC; ++cc) {
                #pragma unroll
                for (int c = 0; c < NC; ++c)
                    av[c] = fmaf(dd[cc], w1S[(2 * NC + cc) * NC + c], av[c]);
            }
            #pragma unroll
            for (int c = 0; c < NC; ++c) vS[tr * NC + c] = (_Float16)av[c];
        }
    }
    __syncthreads();

    // ---------------- phase 2: WMMA over all (j,k); wave wv owns j-tile wv
    const int lane = t & 31;
    const int wv   = t >> 5;       // 0..7 -> j tile
    const int m    = lane & 15;    // A row (pair) / B,C,D column
    const int kh   = lane >> 4;    // which K half this lane carries

    // B fragment: lane (m,kh) holds W2[K][m] for K = 8*kh..8*kh+7 in elems
    // 0..7 (VGPRs 0-3); elems 8..15 (K = 16..31 pad) stay zero.
    v16h Bf = {};
    #pragma unroll
    for (int r = 0; r < 8; ++r) Bf[r] = (_Float16)W2[(8 * kh + r) * NC + m];

    // C init = b2 broadcast down the lane's column (stays in place as the
    // WMMA C operand; no per-iteration copies).
    const float b2v = b2[m];
    v8f cinit;
    #pragma unroll
    for (int r = 0; r < 8; ++r) cinit[r] = b2v;

    const bool isMax = (m < 8);    // cols 0-7: max-reduce, 8-15: mean-reduce
    float accM = 0.0f;             // running max  (relu folded: accM >= 0)
    float accS = 0.0f;             // running sum of relu

    // Hoist this lane's u fragment (channels 8*kh..8*kh+7): one b128 load.
    const int j0 = wv * 16;
    const v8h ua = *(const v8h*)&uS[(j0 + m) * NC + 8 * kh];
    const v8h zero8 = {};

    #pragma unroll 4
    for (int k = 0; k < NK; ++k) {
        const v8h va = *(const v8h*)&vS[k * NC + 8 * kh];
        // relu#1 in packed f16: v_pk_add_f16 + v_pk_max_num_f16
        const v8h h = __builtin_elementwise_max(ua + va, zero8);

        v16h Af = {};              // elems 8..15 remain zero (K padding)
        #pragma unroll
        for (int e = 0; e < 8; ++e) Af[e] = h[e];

        // D[m][n] = b2[n] + sum_K relu1[m][K] * W2[K][n]
        v8f d = __builtin_amdgcn_wmma_f32_16x16x32_f16(
            false, Af, false, Bf, (short)0, cinit, false, false);

        // Tree reductions (no serial per-row chain).
        // max path: accM >= 0, so relu is absorbed by the max.
        float m0 = fmaxf(d[0], d[1]), m1 = fmaxf(d[2], d[3]);
        float m2 = fmaxf(d[4], d[5]), m3 = fmaxf(d[6], d[7]);
        accM = fmaxf(accM, fmaxf(fmaxf(m0, m1), fmaxf(m2, m3)));
        // sum path: relu each row, log-depth add tree, one serial add.
        float r0 = fmaxf(d[0], 0.0f) + fmaxf(d[1], 0.0f);
        float r1 = fmaxf(d[2], 0.0f) + fmaxf(d[3], 0.0f);
        float r2 = fmaxf(d[4], 0.0f) + fmaxf(d[5], 0.0f);
        float r3 = fmaxf(d[6], 0.0f) + fmaxf(d[7], 0.0f);
        accS += (r0 + r1) + (r2 + r3);
    }

    // Combine the two row-halves (rows 0-7 vs 8-15 live in lane^16),
    // then pick the statistic this column needs.
    accM = fmaxf(accM, __shfl_xor(accM, 16, 32));
    accS += __shfl_xor(accS, 16, 32);
    const float tot = isMax ? accM : accS;
    if (lane < 16) redS[wv * NC + m] = tot;
    __syncthreads();

    // Combine the 8 waves and emit [max(8) | mean(8)].
    if (t < NC) {
        float r = redS[t];
        if (t < 8) {
            #pragma unroll
            for (int w = 1; w < 8; ++w) r = fmaxf(r, redS[w * NC + t]);
            out[blk * NC + t] = r;
        } else {
            #pragma unroll
            for (int w = 1; w < 8; ++w) r += redS[w * NC + t];
            out[blk * NC + t] = r * (1.0f / (float)(NJ * NK));
        }
    }
}

extern "C" void kernel_launch(void* const* d_in, const int* in_sizes, int n_in,
                              void* d_out, int out_size, void* d_ws, size_t ws_size,
                              hipStream_t stream) {
    const float* x  = (const float*)d_in[0];
    const float* x1 = (const float*)d_in[1];
    const float* x2 = (const float*)d_in[2];
    const float* W1 = (const float*)d_in[3];
    const float* b1 = (const float*)d_in[4];
    const float* W2 = (const float*)d_in[5];
    const float* b2 = (const float*)d_in[6];
    float* out = (float*)d_out;

    const int nblocks = in_sizes[0] / NC;   // B*N = 512
    pppr_kernel<<<nblocks, 256, 0, stream>>>(x, x1, x2, W1, b1, W2, b2, out);
}
